// AuroraAttention_53429393162822
// MI455X (gfx1250) — matrix operand order repaired and checked by reading, not compiled, not run on hardware
//
#include <hip/hip_runtime.h>

#define E  1024
#define S  2048
#define BB 2
#define H  16
#define Dh 64
#define BS (BB*S)

typedef unsigned int u32;
typedef __attribute__((ext_vector_type(4)))  u32    u32x4;
typedef __attribute__((ext_vector_type(16))) __bf16 v16bf;
typedef __attribute__((ext_vector_type(8)))  float  v8f;

__device__ __forceinline__ v8f wmma_bf16(v16bf a, v16bf b, v8f c) {
    return __builtin_amdgcn_wmma_f32_16x16x32_bf16(false, a, false, b, (short)0, c, false, false);
}

// Two 16-byte loads -> one 16-bf16 fragment lane slice.
__device__ __forceinline__ v16bf load_frag_2x16(const __bf16* p0, const __bf16* p1) {
    union { u32x4 q[2]; v16bf v; } u;
    u.q[0] = *reinterpret_cast<const u32x4*>(p0);
    u.q[1] = *reinterpret_cast<const u32x4*>(p1);
    return u.v;
}
__device__ __forceinline__ v16bf load_frag_32B(const __bf16* p) {
    return load_frag_2x16(p, p + 8);
}

// ---------------------------------------------------------------------------
// f32 -> bf16 conversion of hidden states
__global__ void cvt_x(const float* __restrict__ X, __bf16* __restrict__ Xb, int n) {
    int i = blockIdx.x * blockDim.x + threadIdx.x;
    if (i < n) Xb[i] = (__bf16)X[i];
}

// ---------------------------------------------------------------------------
// Pack nn.Linear weight W (E x E, y = x @ W^T) into WMMA B-fragment tiles.
// Logical B-matrix: Bmat[k][n] = W[n*E + k].
// Packed layout: tile(kt,nt) of 32x16, 512 bf16 per tile; within a tile lane l
// holds 16 K values (K = kt*32 + (l>>4)*8 + (t&7) + (t>>3)*16) of column nt*16+(l&15).
__global__ void pack_w(const float* __restrict__ W, __bf16* __restrict__ out) {
    int idx = blockIdx.x * blockDim.x + threadIdx.x;   // < E*E
    if (idx >= E * E) return;
    int t    = idx & 15;
    int lane = (idx >> 4) & 31;
    int tile = idx >> 9;
    int nt   = tile & (E / 16 - 1);
    int kt   = tile / (E / 16);
    int n    = nt * 16 + (lane & 15);
    int k    = kt * 32 + ((lane >> 4) << 3) + (t & 7) + ((t >> 3) << 4);
    out[idx] = (__bf16)W[n * E + k];
}

// ---------------------------------------------------------------------------
// QKV projection GEMM: (BS x E) @ (E x E) + bias, three outputs selected by z.
// Each wave computes a 16x64 output strip. Branch-free ping-pong pipeline:
// main loop reloads kt+2/kt+3 unconditionally; last two steps are peeled.
__global__ __launch_bounds__(128) void gemm_qkv(
    const __bf16* __restrict__ Xb,
    const __bf16* __restrict__ Wq_pk, const __bf16* __restrict__ Wk_pk, const __bf16* __restrict__ Wv_pk,
    const float* __restrict__ bq, const float* __restrict__ bk, const float* __restrict__ bv,
    __bf16* __restrict__ Qb, __bf16* __restrict__ Kb, __bf16* __restrict__ Vt)
{
    const int lane = threadIdx.x & 31;
    const int wave = threadIdx.x >> 5;
    const int mt   = blockIdx.x * 4 + wave;   // 0..255
    const int n0   = blockIdx.y * 64;
    const int z    = blockIdx.z;              // 0=Q 1=K 2=V
    const __bf16* Wpk  = (z == 0) ? Wq_pk : (z == 1) ? Wk_pk : Wv_pk;
    const float*  bias = (z == 0) ? bq    : (z == 1) ? bk    : bv;

    const int lm = lane & 15, lh = lane >> 4;
    const int arow = mt * 16 + lm;

    const __bf16* abase = Xb + arow * E + lh * 8;
    const __bf16* bbase = Wpk + (n0 >> 4) * 512 + lane * 16;
    const int bstride = (E / 16) * 512;       // packed elements per kt step

    auto loadAB = [&](int kt, v16bf& a, v16bf* bfr) {
        const __bf16* ap = abase + kt * 32;
        a = load_frag_2x16(ap, ap + 16);
        const __bf16* bp = bbase + kt * bstride;
#pragma unroll
        for (int jj = 0; jj < 4; ++jj) bfr[jj] = load_frag_32B(bp + jj * 512);
    };

    v8f acc[4] = {};
    v16bf aA, aB, bA[4], bB[4];
    loadAB(0, aA, bA);
    loadAB(1, aB, bB);

    for (int kt = 0; kt < E / 32 - 2; kt += 2) {    // branch-free steady state
#pragma unroll
        for (int jj = 0; jj < 4; ++jj) acc[jj] = wmma_bf16(aA, bA[jj], acc[jj]);
        loadAB(kt + 2, aA, bA);
#pragma unroll
        for (int jj = 0; jj < 4; ++jj) acc[jj] = wmma_bf16(aB, bB[jj], acc[jj]);
        loadAB(kt + 3, aB, bB);
    }
#pragma unroll
    for (int jj = 0; jj < 4; ++jj) acc[jj] = wmma_bf16(aA, bA[jj], acc[jj]);
#pragma unroll
    for (int jj = 0; jj < 4; ++jj) acc[jj] = wmma_bf16(aB, bB[jj], acc[jj]);

#pragma unroll
    for (int j = 0; j < 4; ++j) {
        int   col = n0 + j * 16 + lm;
        float bv_ = bias[col];
#pragma unroll
        for (int r = 0; r < 8; ++r) {
            int   rr = mt * 16 + r + lh * 8;  // C-layout: M = r + 8*(lane>>4)
            float v  = acc[j][r] + bv_;
            if (z == 2) {
                int b_ = rr >> 11, s_ = rr & (S - 1);
                int h_ = col >> 6, d_ = col & (Dh - 1);
                Vt[(((b_ * H + h_) * Dh + d_) << 11) + s_] = (__bf16)v;
            } else if (z == 0) {
                Qb[rr * E + col] = (__bf16)v;
            } else {
                Kb[rr * E + col] = (__bf16)v;
            }
        }
    }
}

// ---------------------------------------------------------------------------
// Flash-attention: one wave per 16-query tile per (b,h). Online softmax over
// S=2048 keys, two 32-key sub-chunks per outer iteration with ping-pong
// K/V/bias register sets. `more` is a compile-time literal at every call site:
// the steady-state loop reloads unconditionally, the last 64 keys are peeled.
__global__ __launch_bounds__(128) void attn_kernel(
    const __bf16* __restrict__ Qb, const __bf16* __restrict__ Kb, const __bf16* __restrict__ Vt,
    const float* __restrict__ bias, __bf16* __restrict__ Ctx)
{
    __shared__ alignas(16) __bf16 plds[4][16 * 32];
    const int lane = threadIdx.x & 31;
    const int wave = threadIdx.x >> 5;
    const int tid  = blockIdx.x * 4 + wave;       // 0..4095
    const int qt   = tid & (S / 16 - 1);
    const int h    = (tid >> 7) & (H - 1);
    const int b    = tid >> 11;
    const int q0   = qt * 16;
    const int lm   = lane & 15, lh = lane >> 4;

    // Q A-fragments for D=64 (two 16x32 k-steps), kept in registers.
    const __bf16* qrow = Qb + (b * S + q0 + lm) * E + h * Dh;
    v16bf qa0 = load_frag_32B(qrow + lh * 16);
    v16bf qa1 = load_frag_32B(qrow + 32 + lh * 16);

    float m[8], l_[8];
    v8f   o[4] = {};
#pragma unroll
    for (int r = 0; r < 8; ++r) { m[r] = -1e30f; l_[r] = 0.f; }

    const float   scale    = 0.125f;                          // 1/sqrt(64)
    const __bf16* Kbase    = Kb + b * S * E + h * Dh;         // + key*E + d
    const __bf16* Vbase    = Vt + (b * H + h) * Dh * S;       // + d*S + s
    const float*  biasbase = bias + (h * S + q0) * S;         // + row*S + key
    __bf16*       ctxrow   = Ctx + (b * S + q0) * E + h * Dh;
    __bf16*       pl       = plds[wave];

    auto loadK = [&](int j, v16bf* k) {
        const __bf16* kp0 = Kbase + (j + lm) * E + lh * 16;
        k[0] = load_frag_32B(kp0);
        k[1] = load_frag_32B(kp0 + 32);
        const __bf16* kp1 = kp0 + 16 * E;
        k[2] = load_frag_32B(kp1);
        k[3] = load_frag_32B(kp1 + 32);
    };
    auto loadV = [&](int j, v16bf* v) {
#pragma unroll
        for (int jj = 0; jj < 4; ++jj)
            v[jj] = load_frag_32B(Vbase + (jj * 16 + lm) * S + j + lh * 16);
    };
    auto loadBias = [&](int j, float* b0, float* b1) {
#pragma unroll
        for (int r = 0; r < 8; ++r) {
            const float* bp = biasbase + (r + lh * 8) * S + j;
            b0[r] = bp[lm];
            b1[r] = bp[16 + lm];
        }
    };

    // Process one 32-key sub-chunk; when `more` (compile-time literal) is true,
    // refill each register set for key offset `jnext` right after its last use.
    auto process = [&](v16bf* kf, v16bf* vf, float* b0, float* b1,
                       int jnext, bool more) {
        if (more) {
            int jp = jnext + 64;                  // clamped unconditional prefetch
            jp = (jp <= S - 64) ? jp : (S - 64);
            __builtin_prefetch(biasbase + (lane & 15) * S + jp + ((lane >> 4) << 4), 0, 0);
        }

        // ---- scores: two 16x16 tiles ----
        v8f c0 = {}, c1 = {};
        c0 = wmma_bf16(qa0, kf[0], c0);
        c0 = wmma_bf16(qa1, kf[1], c0);
        c1 = wmma_bf16(qa0, kf[2], c1);
        c1 = wmma_bf16(qa1, kf[3], c1);
        if (more) loadK(jnext, kf);               // K set free: refill

        // ---- scale + bias ----
        float t0[8], t1[8];
#pragma unroll
        for (int r = 0; r < 8; ++r) {
            t0[r] = c0[r] * scale + b0[r];
            t1[r] = c1[r] * scale + b1[r];
        }
        if (more) loadBias(jnext, b0, b1);        // bias set free: refill

        // ---- online softmax (reductions over 16-lane halves, wave32) ----
        float pm[8];
#pragma unroll
        for (int r = 0; r < 8; ++r) pm[r] = fmaxf(t0[r], t1[r]);
#pragma unroll
        for (int off = 1; off < 16; off <<= 1)
#pragma unroll
            for (int r = 0; r < 8; ++r) pm[r] = fmaxf(pm[r], __shfl_xor(pm[r], off, 32));

        float alpha[8];
#pragma unroll
        for (int r = 0; r < 8; ++r) {
            float mn = fmaxf(m[r], pm[r]);
            alpha[r] = __expf(m[r] - mn);
            m[r]     = mn;
            t0[r]    = __expf(t0[r] - mn);
            t1[r]    = __expf(t1[r] - mn);
        }
        float ps[8];
#pragma unroll
        for (int r = 0; r < 8; ++r) ps[r] = t0[r] + t1[r];
#pragma unroll
        for (int off = 1; off < 16; off <<= 1)
#pragma unroll
            for (int r = 0; r < 8; ++r) ps[r] += __shfl_xor(ps[r], off, 32);
#pragma unroll
        for (int r = 0; r < 8; ++r) l_[r] = l_[r] * alpha[r] + ps[r];
#pragma unroll
        for (int jj = 0; jj < 4; ++jj)
#pragma unroll
            for (int r = 0; r < 8; ++r) o[jj][r] *= alpha[r];

        // ---- C-layout -> A-layout for P via per-wave LDS tile (16x32 bf16) ----
#pragma unroll
        for (int r = 0; r < 8; ++r) {
            int row = r + lh * 8;
            pl[row * 32 + lm]      = (__bf16)t0[r];
            pl[row * 32 + 16 + lm] = (__bf16)t1[r];
        }
        __syncthreads();
        v16bf pa = load_frag_32B(pl + lm * 32 + lh * 16);
        __syncthreads();

        // ---- P @ V ----
#pragma unroll
        for (int jj = 0; jj < 4; ++jj)
            o[jj] = wmma_bf16(pa, vf[jj], o[jj]);
        if (more) loadV(jnext, vf);               // V set free: refill
    };

    v16bf kA[4], vA[4], kB[4], vB[4];
    float bA0[8], bA1[8], bB0[8], bB1[8];
    loadK(0, kA);  loadV(0, vA);  loadBias(0, bA0, bA1);
    loadK(32, kB); loadV(32, vB); loadBias(32, bB0, bB1);

    for (int j = 0; j < S - 64; j += 64) {        // steady state: reloads always legal
        process(kA, vA, bA0, bA1, j + 64, true);  // keys j    .. j+31
        process(kB, vB, bB0, bB1, j + 96, true);  // keys j+32 .. j+63
    }
    process(kA, vA, bA0, bA1, 0, false);          // keys S-64 .. S-33 (peeled)
    process(kB, vB, bB0, bB1, 0, false);          // keys S-32 .. S-1  (peeled)

    // ---- normalize and store context (bf16, row-major merged heads) ----
#pragma unroll
    for (int jj = 0; jj < 4; ++jj)
#pragma unroll
        for (int r = 0; r < 8; ++r) {
            int row = r + lh * 8;
            ctxrow[row * E + jj * 16 + lm] = (__bf16)(o[jj][r] / l_[r]);
        }
}

// ---------------------------------------------------------------------------
// Output projection: Ctx(bf16) @ Wo^T -> f32 d_out. Same branch-free pipeline.
__global__ __launch_bounds__(128) void gemm_out(
    const __bf16* __restrict__ Ctx, const __bf16* __restrict__ Wpk, float* __restrict__ Out)
{
    const int lane = threadIdx.x & 31;
    const int wave = threadIdx.x >> 5;
    const int mt   = blockIdx.x * 4 + wave;
    const int n0   = blockIdx.y * 64;
    const int lm = lane & 15, lh = lane >> 4;
    const int arow = mt * 16 + lm;

    const __bf16* abase = Ctx + arow * E + lh * 8;
    const __bf16* bbase = Wpk + (n0 >> 4) * 512 + lane * 16;
    const int bstride = (E / 16) * 512;

    auto loadAB = [&](int kt, v16bf& a, v16bf* bfr) {
        const __bf16* ap = abase + kt * 32;
        a = load_frag_2x16(ap, ap + 16);
        const __bf16* bp = bbase + kt * bstride;
#pragma unroll
        for (int jj = 0; jj < 4; ++jj) bfr[jj] = load_frag_32B(bp + jj * 512);
    };

    v8f acc[4] = {};
    v16bf aA, aB, bA[4], bB[4];
    loadAB(0, aA, bA);
    loadAB(1, aB, bB);

    for (int kt = 0; kt < E / 32 - 2; kt += 2) {
#pragma unroll
        for (int jj = 0; jj < 4; ++jj) acc[jj] = wmma_bf16(aA, bA[jj], acc[jj]);
        loadAB(kt + 2, aA, bA);
#pragma unroll
        for (int jj = 0; jj < 4; ++jj) acc[jj] = wmma_bf16(aB, bB[jj], acc[jj]);
        loadAB(kt + 3, aB, bB);
    }
#pragma unroll
    for (int jj = 0; jj < 4; ++jj) acc[jj] = wmma_bf16(aA, bA[jj], acc[jj]);
#pragma unroll
    for (int jj = 0; jj < 4; ++jj) acc[jj] = wmma_bf16(aB, bB[jj], acc[jj]);

#pragma unroll
    for (int j = 0; j < 4; ++j) {
        int col = n0 + j * 16 + lm;
#pragma unroll
        for (int r = 0; r < 8; ++r) {
            int rr = mt * 16 + r + lh * 8;
            Out[rr * E + col] = acc[j][r];
        }
    }
}

// ---------------------------------------------------------------------------
extern "C" void kernel_launch(void* const* d_in, const int* in_sizes, int n_in,
                              void* d_out, int out_size, void* d_ws, size_t ws_size,
                              hipStream_t stream)
{
    (void)in_sizes; (void)n_in; (void)out_size; (void)ws_size;
    const float* hs   = (const float*)d_in[0];
    const float* bias = (const float*)d_in[1];
    const float* Wq   = (const float*)d_in[2];
    const float* bq   = (const float*)d_in[3];
    const float* Wk   = (const float*)d_in[4];
    const float* bk   = (const float*)d_in[5];
    const float* Wv   = (const float*)d_in[6];
    const float* bv   = (const float*)d_in[7];
    const float* Wo   = (const float*)d_in[8];

    char*   ws  = (char*)d_ws;
    __bf16* Xb  = (__bf16*)(ws);                       // 8 MiB
    __bf16* Wqp = (__bf16*)(ws + (8u  << 20));         // 4 x 2 MiB packed weights
    __bf16* Wkp = Wqp + E * E;
    __bf16* Wvp = Wkp + E * E;
    __bf16* Wop = Wvp + E * E;
    __bf16* Qb  = (__bf16*)(ws + (16u << 20));         // 8 MiB
    __bf16* Kb  = (__bf16*)(ws + (24u << 20));         // 8 MiB
    __bf16* Vt  = (__bf16*)(ws + (32u << 20));         // 8 MiB  (per-head transposed)
    __bf16* Ctx = (__bf16*)(ws + (40u << 20));         // 8 MiB

    cvt_x <<<(BS * E) / 256, 256, 0, stream>>>(hs, Xb, BS * E);
    pack_w<<<(E * E) / 256, 256, 0, stream>>>(Wq, Wqp);
    pack_w<<<(E * E) / 256, 256, 0, stream>>>(Wk, Wkp);
    pack_w<<<(E * E) / 256, 256, 0, stream>>>(Wv, Wvp);
    pack_w<<<(E * E) / 256, 256, 0, stream>>>(Wo, Wop);

    dim3 g1(64, 16, 3);
    gemm_qkv<<<g1, 128, 0, stream>>>(Xb, Wqp, Wkp, Wvp, bq, bk, bv, Qb, Kb, Vt);

    attn_kernel<<<(BB * H * (S / 16)) / 4, 128, 0, stream>>>(Qb, Kb, Vt, bias, Ctx);

    dim3 g2(64, 16, 1);
    gemm_out<<<g2, 128, 0, stream>>>(Ctx, Wop, (float*)d_out);
}
